// TransformerLayer_26474178412771
// MI455X (gfx1250) — matrix operand order repaired and checked by reading, not compiled, not run on hardware
//
#include <hip/hip_runtime.h>
#include <hip/hip_bf16.h>
#include <cmath>
#include <cstdint>

// ---------------------------------------------------------------------------
// MI455X (gfx1250, wave32) implementation of the two-stream transformer layer.
// All matmuls (projections, attention QK^T / PV, FFN) run on
// v_wmma_f32_16x16x32_bf16 with fp32 accumulate (~0.45 TFLOP total =>
// compute-bound vs ~13us of HBM traffic at 23.3 TB/s). GEMM staging uses the
// CDNA5 async DMA path (global_load_async_to_lds_b128, ASYNCcnt) with
// double-buffered LDS so the next K-slab fetch overlaps the current WMMAs.
// ---------------------------------------------------------------------------

#define B_  2
#define N_  2048
#define D_  1024
#define H_  16
#define HD_ 64
#define M_  (B_ * N_)   // 4096 flattened rows

typedef __attribute__((ext_vector_type(16))) __bf16 bf16x16;
typedef __attribute__((ext_vector_type(8)))  float  v8f;

union FragU { bf16x16 v; uint32_t u[8]; };

__device__ inline v8f vzero() {
    v8f z;
#pragma unroll
    for (int i = 0; i < 8; ++i) z[i] = 0.0f;
    return z;
}

__device__ inline v8f wmma_bf16(bf16x16 a, bf16x16 b, v8f c) {
    return __builtin_amdgcn_wmma_f32_16x16x32_bf16(
        /*neg_a=*/false, a, /*neg_b=*/false, b,
        /*c_mod=*/(short)0, c, /*reuse_a=*/false, /*reuse_b=*/false);
}

// Workgroup-relative LDS byte offset (low 32 bits of the generic pointer).
__device__ inline uint32_t lds_addr32(const void* p) {
    return (uint32_t)(uintptr_t)p;
}

// A-matrix fragment (16x32 bf16). ISA layout: lanes 0-15 carry M=lane,
// lanes 16-31 carry same M with K offset by 8 in the upper half:
//   dword d<4 : K = half*8 + 2d   ; dword d>=4 : K = 16 + half*8 + 2(d-4)
// `row` points at the start of this lane's M-row; K indexed within the row.
__device__ inline bf16x16 load_afrag(const __bf16* row, int half) {
    FragU f;
#pragma unroll
    for (int d = 0; d < 8; ++d) {
        int k0 = (d < 4) ? (half * 8 + 2 * d) : (16 + half * 8 + 2 * (d - 4));
        f.u[d] = *(const uint32_t*)(row + k0);   // K0 even -> dword aligned
    }
    return f.v;
}

// B-matrix fragment (32x16 bf16). Lane l holds column N=l%16; element s
// holds K = half*8 + s (s<8) or 16 + half*8 + (s-8) (s>=8), same as A.
// `colptr` = &B[K=0][n0 + lane16], `ld` = row stride of B.
__device__ inline bf16x16 load_bfrag(const __bf16* colptr, int ld, int half) {
    bf16x16 v;
#pragma unroll
    for (int s = 0; s < 16; ++s) {
        int k = (s < 8) ? (half * 8 + s) : (16 + half * 8 + (s - 8));
        v[s] = colptr[(size_t)k * ld];
    }
    return v;
}

// ---------------------------------------------------------------------------
// Generic bf16 GEMM: C[M,N] = A[M,K] * B[K,N] + bias (+resid), M = 4096.
// Block: 256 threads = 8 waves; block tile 128(M) x 64(N); wave tile 32x32
// (2x2 WMMA tiles). K staged in steps of 32 via async DMA into
// double-buffered LDS: stage(k+1) overlaps compute(k); s_wait_asynccnt 3
// (loads retire in order) guarantees the current slab has landed.
// ---------------------------------------------------------------------------
__global__ __launch_bounds__(256) void k_gemm(
    const __bf16* __restrict__ A, int lda,
    const __bf16* __restrict__ Bw, int ldb,
    const float*  __restrict__ bias,
    const float*  __restrict__ resid, int ldr,
    float*        __restrict__ outF,  int ldo,
    __bf16*       __restrict__ outH,  int ldh, int hcol0,
    int Nn, int Kk)
{
    __shared__ __align__(16) __bf16 As[2][128][32];
    __shared__ __align__(16) __bf16 Bs[2][32][64];

    const int tid    = threadIdx.x;
    const int lane   = tid & 31;
    const int wid    = tid >> 5;
    const int waveM  = wid >> 1;          // 0..3
    const int waveN  = wid & 1;           // 0..1
    const int lane16 = lane & 15;
    const int half   = lane >> 4;
    const int rowBase = blockIdx.y * 128;
    const int colBase = blockIdx.x * 64;

    // cooperative staging indices
    const int arow = tid >> 1;            // 0..127
    const int acol = (tid & 1) * 16;      // 0 / 16
    const int brow = tid >> 3;            // 0..31
    const int bcol = (tid & 7) * 8;       // 0..56

    const __bf16* aG = A  + (size_t)(rowBase + arow) * lda + acol;
    const __bf16* bG = Bw + (size_t)brow * ldb + colBase + bcol;

    // Issue one K-slab of async global->LDS DMA (3 instrs / thread):
    // 16 bf16 of A (2 x b128, offset applies to both LDS and global addr)
    // and 8 bf16 of B (1 x b128).
    auto stage = [&](int k0, int buf) {
        uint32_t al = lds_addr32(&As[buf][arow][acol]);
        uint64_t ag = (uint64_t)(uintptr_t)(aG + k0);
        uint32_t bl = lds_addr32(&Bs[buf][brow][bcol]);
        uint64_t bg = (uint64_t)(uintptr_t)(bG + (size_t)k0 * ldb);
        asm volatile("global_load_async_to_lds_b128 %0, %1, off"
                     :: "v"(al), "v"(ag) : "memory");
        asm volatile("global_load_async_to_lds_b128 %0, %1, off offset:16"
                     :: "v"(al), "v"(ag) : "memory");
        asm volatile("global_load_async_to_lds_b128 %0, %1, off"
                     :: "v"(bl), "v"(bg) : "memory");
    };

    v8f acc[2][2];
#pragma unroll
    for (int i = 0; i < 2; ++i)
#pragma unroll
        for (int j = 0; j < 2; ++j) acc[i][j] = vzero();

    stage(0, 0);   // prologue

    for (int k0 = 0; k0 < Kk; k0 += 32) {
        const int cur = (k0 >> 5) & 1;
        if (k0 + 32 < Kk) {
            stage(k0 + 32, cur ^ 1);                       // overlap next fetch
            asm volatile("s_wait_asynccnt 0x3" ::: "memory");  // cur slab landed
        } else {
            asm volatile("s_wait_asynccnt 0x0" ::: "memory");
        }
        __syncthreads();   // all waves' DMA visible

        bf16x16 af[2], bf[2];
#pragma unroll
        for (int i = 0; i < 2; ++i)
            af[i] = load_afrag(&As[cur][waveM * 32 + i * 16 + lane16][0], half);
#pragma unroll
        for (int j = 0; j < 2; ++j)
            bf[j] = load_bfrag(&Bs[cur][0][waveN * 32 + j * 16 + lane16], 64, half);
#pragma unroll
        for (int i = 0; i < 2; ++i)
#pragma unroll
            for (int j = 0; j < 2; ++j)
                acc[i][j] = wmma_bf16(af[i], bf[j], acc[i][j]);
        __syncthreads();   // reads done before next iter's DMA overwrites buf
    }

    // epilogue: C layout -> VGPR r holds M = r + half*8, N = lane16
#pragma unroll
    for (int i = 0; i < 2; ++i)
#pragma unroll
        for (int j = 0; j < 2; ++j)
#pragma unroll
            for (int r = 0; r < 8; ++r) {
                int row = rowBase + waveM * 32 + i * 16 + half * 8 + r;
                int col = colBase + waveN * 32 + j * 16 + lane16;
                float v = acc[i][j][r] + bias[col];
                if (resid) v += resid[(size_t)row * ldr + col];
                if (outF)  outF[(size_t)row * ldo + col] = v;
                if (outH)  outH[(size_t)row * ldh + hcol0 + col] = (__bf16)v;
            }
}

// ---------------------------------------------------------------------------
// Flash attention: one wave per 16-row Q tile, 8 waves per block (128 rows).
// Per 32-key step: 4 WMMA for S = Q*K^T (K pre-packed dim-major), online
// softmax (shfl_xor row reductions), P staged through per-wave LDS tile to
// re-lay into an A-fragment (guarded by s_wait_dscnt), 4 WMMA for P*V.
// ---------------------------------------------------------------------------
__global__ __launch_bounds__(256) void k_flash(
    const __bf16* __restrict__ Qh,   // [B,H,N,HD]
    const __bf16* __restrict__ Kt,   // [B,H,HD,N]
    const __bf16* __restrict__ Vh,   // [B,H,N,HD]
    __bf16*       __restrict__ ctx)  // [B,N,D] merged heads
{
    __shared__ __align__(16) __bf16 Pst[8][16][32];

    const int tid    = threadIdx.x;
    const int lane   = tid & 31;
    const int w      = tid >> 5;
    const int lane16 = lane & 15;
    const int half   = lane >> 4;
    const int nblk   = N_ / 128;                 // 16
    const int bh     = blockIdx.x / nblk;
    const int qb     = (blockIdx.x % nblk) * 128 + w * 16;
    const int b      = bh / H_;
    const int h      = bh % H_;

    const __bf16* qrow = Qh + ((size_t)bh * N_ + qb + lane16) * HD_;
    bf16x16 aq0 = load_afrag(qrow,      half);   // dims 0..31
    bf16x16 aq1 = load_afrag(qrow + 32, half);   // dims 32..63

    v8f o[4];
#pragma unroll
    for (int c = 0; c < 4; ++c) o[c] = vzero();
    float mrun[8], lrun[8];
#pragma unroll
    for (int r = 0; r < 8; ++r) { mrun[r] = -INFINITY; lrun[r] = 0.0f; }

    const float scale = 0.125f;  // 1/sqrt(64)

    for (int kb = 0; kb < N_; kb += 32) {
        v8f s[2];
#pragma unroll
        for (int g = 0; g < 2; ++g) {
            bf16x16 b0 = load_bfrag(Kt + ((size_t)bh * HD_)      * N_ + kb + g * 16 + lane16, N_, half);
            bf16x16 b1 = load_bfrag(Kt + ((size_t)bh * HD_ + 32) * N_ + kb + g * 16 + lane16, N_, half);
            v8f t = vzero();
            t = wmma_bf16(aq0, b0, t);
            t = wmma_bf16(aq1, b1, t);
            s[g] = t;
        }
#pragma unroll
        for (int r = 0; r < 8; ++r) {
            float s0 = s[0][r] * scale, s1 = s[1][r] * scale;
            float tm = fmaxf(s0, s1);
#pragma unroll
            for (int mkl = 1; mkl < 16; mkl <<= 1)
                tm = fmaxf(tm, __shfl_xor(tm, mkl, 32));   // masks<16: stays in half
            float mn = fmaxf(mrun[r], tm);
            float alpha = __expf(mrun[r] - mn);
            float p0 = __expf(s0 - mn), p1 = __expf(s1 - mn);
            float rs = p0 + p1;
#pragma unroll
            for (int mkl = 1; mkl < 16; mkl <<= 1)
                rs += __shfl_xor(rs, mkl, 32);
            lrun[r] = lrun[r] * alpha + rs;
            mrun[r] = mn;
#pragma unroll
            for (int c = 0; c < 4; ++c) o[c][r] *= alpha;
            int mrow = half * 8 + r;
            Pst[w][mrow][lane16]      = (__bf16)p0;
            Pst[w][mrow][16 + lane16] = (__bf16)p1;
        }
        asm volatile("s_wait_dscnt 0x0" ::: "memory");   // wave-local LDS RAW
        bf16x16 ap = load_afrag(&Pst[w][lane16][0], half);
#pragma unroll
        for (int c = 0; c < 4; ++c) {
            bf16x16 bv = load_bfrag(Vh + ((size_t)bh * N_ + kb) * HD_ + c * 16 + lane16, HD_, half);
            o[c] = wmma_bf16(ap, bv, o[c]);
        }
    }

#pragma unroll
    for (int r = 0; r < 8; ++r) {
        float inv = 1.0f / lrun[r];
        int row = qb + half * 8 + r;
#pragma unroll
        for (int c = 0; c < 4; ++c) {
            float val = o[c][r] * inv;
            ctx[((size_t)b * N_ + row) * D_ + h * HD_ + c * 16 + lane16] = (__bf16)val;
        }
    }
}

// ---------------------------------------------------------------------------
// Elementwise / packing kernels
// ---------------------------------------------------------------------------
__global__ void k_f32_to_bf16(const float* __restrict__ s, __bf16* __restrict__ d, long long n)
{
    for (long long i = (long long)blockIdx.x * blockDim.x + threadIdx.x;
         i < n; i += (long long)gridDim.x * blockDim.x)
        d[i] = (__bf16)s[i];
}

__global__ void k_f32_to_bf16_strided(const float* __restrict__ s,
                                      __bf16* __restrict__ d, int cols, int dstld)
{
    long long i = (long long)blockIdx.x * blockDim.x + threadIdx.x;
    long long row = i / cols, col = i % cols;
    d[(size_t)row * dstld + col] = (__bf16)s[i];
}

// qkv fp32 [B,N,3D], inner dim ordered (h, hd, 3). Applies RoPE to q,k and
// packs Qh/Vh [B,H,N,HD] + Kt [B,H,HD,N] in bf16.
__global__ void k_pack_qkv_rope(const float* __restrict__ qkv,
                                const float* __restrict__ enc,  // [2,B,1,N,HD]
                                __bf16* __restrict__ Qh, __bf16* __restrict__ Kt,
                                __bf16* __restrict__ Vh)
{
    int idx = blockIdx.x * blockDim.x + threadIdx.x;   // < B*N*H*HD
    int d  = idx % HD_;
    int hh = (idx / HD_) % H_;
    int n  = (idx / (HD_ * H_)) % N_;
    int b  = idx / (HD_ * H_ * N_);
    size_t base  = (size_t)(b * N_ + n) * (3 * D_) + (hh * HD_ + d) * 3;
    size_t basep = (size_t)(b * N_ + n) * (3 * D_) + (hh * HD_ + (d ^ 1)) * 3;
    float q = qkv[base + 0], k = qkv[base + 1], v = qkv[base + 2];
    float qp = qkv[basep + 0], kp = qkv[basep + 1];
    float sgn = (d & 1) ? 1.0f : -1.0f;   // rot[2i]=-t[2i+1], rot[2i+1]=t[2i]
    float f0 = enc[((size_t)(0 * B_ + b) * N_ + n) * HD_ + d];
    float f1 = enc[((size_t)(1 * B_ + b) * N_ + n) * HD_ + d];
    float qr = q * f0 + sgn * qp * f1;
    float kr = k * f0 + sgn * kp * f1;
    size_t qo = ((size_t)(b * H_ + hh) * N_ + n) * HD_ + d;
    Qh[qo] = (__bf16)qr;
    Vh[qo] = (__bf16)v;
    Kt[((size_t)(b * H_ + hh) * HD_ + d) * N_ + n] = (__bf16)kr;
}

// x fp32 [B,N,D] -> any of Qh [B,H,N,HD], Kt [B,H,HD,N], Vh [B,H,N,HD] (bf16)
__global__ void k_pack_heads(const float* __restrict__ x,
                             __bf16* Qh, __bf16* Kt, __bf16* Vh)
{
    int idx = blockIdx.x * blockDim.x + threadIdx.x;
    int d  = idx % HD_;
    int hh = (idx / HD_) % H_;
    int n  = (idx / (HD_ * H_)) % N_;
    int b  = idx / (HD_ * H_ * N_);
    float t = x[(size_t)(b * N_ + n) * D_ + hh * HD_ + d];
    size_t qo = ((size_t)(b * H_ + hh) * N_ + n) * HD_ + d;
    if (Qh) Qh[qo] = (__bf16)t;
    if (Vh) Vh[qo] = (__bf16)t;
    if (Kt) Kt[((size_t)(b * H_ + hh) * HD_ + d) * N_ + n] = (__bf16)t;
}

// LayerNorm (population var, eps 1e-5) + exact GELU; row length = cols.
__global__ __launch_bounds__(256) void k_lngelu(
    const float* __restrict__ x, const float* __restrict__ g,
    const float* __restrict__ bta, __bf16* __restrict__ out, int cols)
{
    __shared__ float red[256];
    int row = blockIdx.x;
    const float* xr = x + (size_t)row * cols;
    float s = 0.0f, s2 = 0.0f;
    for (int c = threadIdx.x; c < cols; c += 256) {
        float v = xr[c]; s += v; s2 += v * v;
    }
    red[threadIdx.x] = s; __syncthreads();
    for (int st = 128; st > 0; st >>= 1) {
        if (threadIdx.x < st) red[threadIdx.x] += red[threadIdx.x + st];
        __syncthreads();
    }
    float mean = red[0] / cols; __syncthreads();
    red[threadIdx.x] = s2; __syncthreads();
    for (int st = 128; st > 0; st >>= 1) {
        if (threadIdx.x < st) red[threadIdx.x] += red[threadIdx.x + st];
        __syncthreads();
    }
    float var  = red[0] / cols - mean * mean;
    float rstd = rsqrtf(var + 1e-5f);
    for (int c = threadIdx.x; c < cols; c += 256) {
        float v  = (xr[c] - mean) * rstd * g[c] + bta[c];
        float ge = 0.5f * v * (1.0f + erff(v * 0.70710678118654752f));
        out[(size_t)row * cols + c] = (__bf16)ge;
    }
}

// ---------------------------------------------------------------------------
// Host orchestration
// ---------------------------------------------------------------------------
extern "C" void kernel_launch(void* const* d_in, const int* in_sizes, int n_in,
                              void* d_out, int out_size, void* d_ws, size_t ws_size,
                              hipStream_t stream)
{
    (void)in_sizes; (void)n_in; (void)out_size; (void)ws_size;
    const float* desc0  = (const float*)d_in[0];
    const float* desc1  = (const float*)d_in[1];
    const float* enc0   = (const float*)d_in[2];
    const float* enc1   = (const float*)d_in[3];
    const float* wqkv_w = (const float*)d_in[4];
    const float* wqkv_b = (const float*)d_in[5];
    const float* sout_w = (const float*)d_in[6];
    const float* sout_b = (const float*)d_in[7];
    const float* sf1_w  = (const float*)d_in[8];
    const float* sf1_b  = (const float*)d_in[9];
    const float* sln_g  = (const float*)d_in[10];
    const float* sln_b  = (const float*)d_in[11];
    const float* sf2_w  = (const float*)d_in[12];
    const float* sf2_b  = (const float*)d_in[13];
    const float* qk_w   = (const float*)d_in[14];
    const float* qk_b   = (const float*)d_in[15];
    const float* v_w    = (const float*)d_in[16];
    const float* v_b    = (const float*)d_in[17];
    const float* cout_w = (const float*)d_in[18];
    const float* cout_b = (const float*)d_in[19];
    const float* cf1_w  = (const float*)d_in[20];
    const float* cf1_b  = (const float*)d_in[21];
    const float* cln_g  = (const float*)d_in[22];
    const float* cln_b  = (const float*)d_in[23];
    const float* cf2_w  = (const float*)d_in[24];
    const float* cf2_b  = (const float*)d_in[25];

    char* ws = (char*)d_ws;
    size_t off = 0;
    auto take = [&](size_t elems, size_t esz) -> void* {
        void* p = ws + off;
        off = (off + elems * esz + 255) & ~(size_t)255;
        return p;
    };
    // bf16 weight copies
    __bf16* Wqkv = (__bf16*)take((size_t)D_ * 3 * D_, 2);
    __bf16* Wso  = (__bf16*)take((size_t)D_ * D_, 2);
    __bf16* Wsf1 = (__bf16*)take((size_t)2 * D_ * 2 * D_, 2);
    __bf16* Wsf2 = (__bf16*)take((size_t)2 * D_ * D_, 2);
    __bf16* Wqk  = (__bf16*)take((size_t)D_ * D_, 2);
    __bf16* Wv   = (__bf16*)take((size_t)D_ * D_, 2);
    __bf16* Wco  = (__bf16*)take((size_t)D_ * D_, 2);
    __bf16* Wcf1 = (__bf16*)take((size_t)2 * D_ * 2 * D_, 2);
    __bf16* Wcf2 = (__bf16*)take((size_t)2 * D_ * D_, 2);
    // scratch
    float*  S1  = (float*)take((size_t)M_ * 3 * D_, 4);   // qkv buf / ffn hidden
    float*  Tf  = (float*)take((size_t)M_ * D_, 4);       // cross proj temp
    float*  D0f = (float*)take((size_t)M_ * D_, 4);       // self-block outputs
    float*  D1f = (float*)take((size_t)M_ * D_, 4);
    __bf16* XB  = (__bf16*)take((size_t)M_ * D_, 2);
    __bf16* P1  = (__bf16*)take((size_t)M_ * D_, 2);      // Qh
    __bf16* P2  = (__bf16*)take((size_t)M_ * D_, 2);      // Kt
    __bf16* P3  = (__bf16*)take((size_t)M_ * D_, 2);      // Vh
    __bf16* P4  = (__bf16*)take((size_t)M_ * D_, 2);
    __bf16* P5  = (__bf16*)take((size_t)M_ * D_, 2);
    __bf16* P6  = (__bf16*)take((size_t)M_ * D_, 2);
    __bf16* CTX = (__bf16*)take((size_t)M_ * D_, 2);
    __bf16* C0  = (__bf16*)take((size_t)M_ * 2 * D_, 2);  // concat buffers
    __bf16* C1  = (__bf16*)take((size_t)M_ * 2 * D_, 2);
    __bf16* HG  = (__bf16*)take((size_t)M_ * 2 * D_, 2);  // gelu(layernorm)
    __bf16* D0B = (__bf16*)take((size_t)M_ * D_, 2);
    __bf16* D1B = (__bf16*)take((size_t)M_ * D_, 2);

    auto conv = [&](const float* s, __bf16* d, long long n) {
        k_f32_to_bf16<<<1024, 256, 0, stream>>>(s, d, n);
    };
    auto gemm = [&](const __bf16* A, int lda, const __bf16* Bw, int ldb,
                    const float* bias, const float* resid, int ldr,
                    float* outF, int ldo, __bf16* outH, int ldh, int hcol0,
                    int Nn, int Kk) {
        dim3 grid(Nn / 64, M_ / 128);
        k_gemm<<<grid, 256, 0, stream>>>(A, lda, Bw, ldb, bias, resid, ldr,
                                         outF, ldo, outH, ldh, hcol0, Nn, Kk);
    };
    auto flash = [&](const __bf16* Q, const __bf16* Kt, const __bf16* V, __bf16* c) {
        k_flash<<<B_ * H_ * (N_ / 128), 256, 0, stream>>>(Q, Kt, V, c);
    };
    const int NE = M_ * D_;        // 4,194,304
    const int GE = NE / 256;

    // weights -> bf16 (recomputed every call; deterministic)
    conv(wqkv_w, Wqkv, (long long)D_ * 3 * D_);
    conv(sout_w, Wso,  (long long)D_ * D_);
    conv(sf1_w,  Wsf1, (long long)2 * D_ * 2 * D_);
    conv(sf2_w,  Wsf2, (long long)2 * D_ * D_);
    conv(qk_w,   Wqk,  (long long)D_ * D_);
    conv(v_w,    Wv,   (long long)D_ * D_);
    conv(cout_w, Wco,  (long long)D_ * D_);
    conv(cf1_w,  Wcf1, (long long)2 * D_ * 2 * D_);
    conv(cf2_w,  Wcf2, (long long)2 * D_ * D_);

    // ---- self blocks ----
    auto self_block = [&](const float* x, const float* enc, float* dOut, __bf16* dOutB) {
        conv(x, XB, NE);
        gemm(XB, D_, Wqkv, 3 * D_, wqkv_b, nullptr, 0, S1, 3 * D_,
             nullptr, 0, 0, 3 * D_, D_);
        k_pack_qkv_rope<<<GE, 256, 0, stream>>>(S1, enc, P1, P2, P3);
        flash(P1, P2, P3, CTX);
        k_f32_to_bf16_strided<<<GE, 256, 0, stream>>>(x, C0, D_, 2 * D_);
        gemm(CTX, D_, Wso, D_, sout_b, nullptr, 0, nullptr, 0,
             C0, 2 * D_, D_, D_, D_);                       // msg -> concat hi half
        gemm(C0, 2 * D_, Wsf1, 2 * D_, sf1_b, nullptr, 0, S1, 2 * D_,
             nullptr, 0, 0, 2 * D_, 2 * D_);
        k_lngelu<<<M_, 256, 0, stream>>>(S1, sln_g, sln_b, HG, 2 * D_);
        gemm(HG, 2 * D_, Wsf2, D_, sf2_b, x, D_, dOut, D_,
             dOutB, D_, 0, D_, 2 * D_);                     // + residual x
    };
    self_block(desc0, enc0, D0f, D0B);
    self_block(desc1, enc1, D1f, D1B);

    // ---- cross block ----
    gemm(D0B, D_, Wqk, D_, qk_b, nullptr, 0, Tf, D_, nullptr, 0, 0, D_, D_);
    k_pack_heads<<<GE, 256, 0, stream>>>(Tf, P1, P2, nullptr);   // qk0 as Q and K^T
    gemm(D1B, D_, Wqk, D_, qk_b, nullptr, 0, Tf, D_, nullptr, 0, 0, D_, D_);
    k_pack_heads<<<GE, 256, 0, stream>>>(Tf, P4, P5, nullptr);   // qk1
    gemm(D0B, D_, Wv, D_, v_b, nullptr, 0, Tf, D_, nullptr, 0, 0, D_, D_);
    k_pack_heads<<<GE, 256, 0, stream>>>(Tf, nullptr, nullptr, P3);  // v0
    gemm(D1B, D_, Wv, D_, v_b, nullptr, 0, Tf, D_, nullptr, 0, 0, D_, D_);
    k_pack_heads<<<GE, 256, 0, stream>>>(Tf, nullptr, nullptr, P6);  // v1

    // m0 = attn(qk0, qk1, v1) @ cout ; concat [d0, m0]
    flash(P1, P5, P6, CTX);
    k_f32_to_bf16_strided<<<GE, 256, 0, stream>>>(D0f, C0, D_, 2 * D_);
    gemm(CTX, D_, Wco, D_, cout_b, nullptr, 0, nullptr, 0, C0, 2 * D_, D_, D_, D_);
    // m1 = attn(qk1, qk0, v0) @ cout ; concat [d1, m1]
    flash(P4, P2, P3, CTX);
    k_f32_to_bf16_strided<<<GE, 256, 0, stream>>>(D1f, C1, D_, 2 * D_);
    gemm(CTX, D_, Wco, D_, cout_b, nullptr, 0, nullptr, 0, C1, 2 * D_, D_, D_, D_);

    float* y0 = (float*)d_out;
    float* y1 = y0 + (size_t)M_ * D_;
    gemm(C0, 2 * D_, Wcf1, 2 * D_, cf1_b, nullptr, 0, S1, 2 * D_,
         nullptr, 0, 0, 2 * D_, 2 * D_);
    k_lngelu<<<M_, 256, 0, stream>>>(S1, cln_g, cln_b, HG, 2 * D_);
    gemm(HG, 2 * D_, Wcf2, D_, cf2_b, D0f, D_, y0, D_, nullptr, 0, 0, D_, 2 * D_);

    gemm(C1, 2 * D_, Wcf1, 2 * D_, cf1_b, nullptr, 0, S1, 2 * D_,
         nullptr, 0, 0, 2 * D_, 2 * D_);
    k_lngelu<<<M_, 256, 0, stream>>>(S1, cln_g, cln_b, HG, 2 * D_);
    gemm(HG, 2 * D_, Wcf2, D_, cf2_b, D1f, D_, y1, D_, nullptr, 0, 0, D_, 2 * D_);
}
